// CARAFE_30219389894859
// MI455X (gfx1250) — matrix operand order repaired and checked by reading, not compiled, not run on hardware
//
#include <hip/hip_runtime.h>
#include <hip/hip_bf16.h>

// CDNA5 WMMA operand vector types (wave32)
typedef __attribute__((ext_vector_type(16))) __bf16 v16bf;
typedef __attribute__((ext_vector_type(8)))  __bf16 v8bf;
typedef __attribute__((ext_vector_type(8)))  float  v8f;

#define BN_EPS 1e-5f

// Geometry: X [4,64,128,128] f32; 1x1 conv 64->64 (+BN,ReLU);
// 3x3 conv 64->100 (+bias,BN); pixel-shuffle r=2 -> 25 softmax planes 256x256;
// CARAFE k=5 dil=2 -> out [4,64,256,256] f32.
//
// Data layout strategy (WMMA-native):
//   Xt    bf16 [b][pix=16384][ci=64]   (K contiguous -> B operand = 1x 32B load)
//   wA1   bf16 [co=64][ci=64]          (A operand = 2x 16B loads)
//   wA2   bf16 [tap=9][co=112][ci=64]  (co zero-padded 100->112)
//   compT bf16 [b][pix][co=64]         (C store = 1x 16B store/lane)
//   encT  bf16 [b][pix][co=112]        (softmax gather hits one 224B record)
//   wm    bf16 [b][k=25][65536]        (apply reads coalesced planes)

// ---------------------------------------------------------------------------
// Kernel 0a: repack weights into bf16 WMMA-A-friendly layouts.
// ---------------------------------------------------------------------------
__global__ __launch_bounds__(256) void k_repack(
    const float* __restrict__ comp_w, const float* __restrict__ enc_w,
    __bf16* __restrict__ wA1, __bf16* __restrict__ wA2)
{
  const int i = blockIdx.x * 256 + threadIdx.x;
  if (i < 4096) wA1[i] = (__bf16)comp_w[i];          // [co][ci], already K-major
  const int j = i - 4096;
  if (j >= 0 && j < 9 * 112 * 64) {
    const int tap = j / (112 * 64);
    const int r   = j - tap * (112 * 64);
    const int co = r >> 6, ci = r & 63;
    float v = 0.0f;
    if (co < 100)
      v = enc_w[((co * 64 + ci) * 3 + tap / 3) * 3 + tap % 3];
    wA2[j] = (__bf16)v;
  }
}

// ---------------------------------------------------------------------------
// Kernel 0b: LDS tile-transpose X (NCHW f32) -> Xt (pixel-major bf16).
// Block handles 64ch x 64pix; both global phases fully coalesced.
// ---------------------------------------------------------------------------
__global__ __launch_bounds__(256) void k_transposeX(
    const float* __restrict__ X, __bf16* __restrict__ Xt)
{
  __shared__ float tile[64 * 65];                 // +1 pad: conflict-free
  const int b  = blockIdx.x >> 8;                 // 256 pixel-tiles of 64
  const int p0 = (blockIdx.x & 255) << 6;
#pragma unroll
  for (int it = 0; it < 16; ++it) {
    const int idx = it * 256 + threadIdx.x;
    const int ch = idx >> 6, px = idx & 63;       // px fastest -> coalesced read
    tile[px * 65 + ch] = X[((b * 64 + ch) << 14) + p0 + px];
  }
  __syncthreads();
#pragma unroll
  for (int it = 0; it < 16; ++it) {
    const int idx = it * 256 + threadIdx.x;
    const int px = idx >> 6, ch = idx & 63;       // ch fastest -> coalesced write
    Xt[((b << 14) + p0 + px) * 64 + ch] = (__bf16)tile[px * 65 + ch];
  }
}

// ---------------------------------------------------------------------------
// Kernel 1: 1x1 conv + BN + ReLU. One wave: 16 pixels x all 64 out-channels,
// B operand (activations) reused across 4 M-tiles. 8 WMMAs/wave.
// ---------------------------------------------------------------------------
__global__ __launch_bounds__(256) void k_compress(
    const __bf16* __restrict__ Xt, const __bf16* __restrict__ wA1,
    const float* __restrict__ g,  const float* __restrict__ be,
    const float* __restrict__ mu, const float* __restrict__ va,
    __bf16* __restrict__ compT)
{
  const int wave = (blockIdx.x << 3) + (threadIdx.x >> 5);
  const int lane = threadIdx.x & 31;
  const int m = lane & 15, half = lane >> 4;
  const int b  = wave >> 10;
  const int p0 = (wave & 1023) << 4;
  const int rec = (b << 14) + p0 + m;             // this lane's pixel record

  v8f acc[4] = {};
#pragma unroll
  for (int kc = 0; kc < 2; ++kc) {
    // B 32x16: lane (m,half), element e -> K = half*16+e, N = m
    const v16bf bm = *(const v16bf*)(Xt + rec * 64 + kc * 32 + half * 16);
#pragma unroll
    for (int mt = 0; mt < 4; ++mt) {
      // A 16x32: element e -> K = half*16+e
      const __bf16* wrow = wA1 + (mt * 16 + m) * 64 + kc * 32;
      const v8bf a0 = *(const v8bf*)(wrow + half * 16);
      const v8bf a1 = *(const v8bf*)(wrow + half * 16 + 8);
      v16bf a;
#pragma unroll
      for (int e = 0; e < 8; ++e) { a[e] = a0[e]; a[8 + e] = a1[e]; }
      acc[mt] = __builtin_amdgcn_wmma_f32_16x16x32_bf16(false, a, false, bm,
                                                        (short)0, acc[mt],
                                                        false, false);
    }
  }
#pragma unroll
  for (int mt = 0; mt < 4; ++mt) {
    v8bf o;
#pragma unroll
    for (int r = 0; r < 8; ++r) {                 // C/D: M = half*8+r, N = m
      const int co = mt * 16 + half * 8 + r;
      const float inv = g[co] * __frsqrt_rn(va[co] + BN_EPS);
      const float v = fmaf(acc[mt][r], inv, be[co] - mu[co] * inv);
      o[r] = (__bf16)fmaxf(v, 0.0f);
    }
    *(v8bf*)(compT + rec * 64 + mt * 16 + half * 8) = o;  // 16B store
  }
}

// ---------------------------------------------------------------------------
// Kernel 2: 3x3 conv (implicit GEMM, K=576) + bias + BN. One wave: 16 pixels x
// all 112 out-channels; B reused across 7 M-tiles. 9 taps x 2 Kc x 7 mt = 126
// WMMAs/wave, all operands 128-bit loads.
// ---------------------------------------------------------------------------
__global__ __launch_bounds__(256) void k_encode(
    const __bf16* __restrict__ compT, const __bf16* __restrict__ wA2,
    const float* __restrict__ bias,
    const float* __restrict__ g,  const float* __restrict__ be,
    const float* __restrict__ mu, const float* __restrict__ va,
    __bf16* __restrict__ encT)
{
  const int wave = (blockIdx.x << 3) + (threadIdx.x >> 5);
  const int lane = threadIdx.x & 31;
  const int m = lane & 15, half = lane >> 4;
  const int b  = wave >> 10;
  const int p0 = (wave & 1023) << 4;
  const int px = (p0 + m) & 127, py = (p0 + m) >> 7;

  v8f acc[7] = {};
#pragma unroll
  for (int tap = 0; tap < 9; ++tap) {
    const int dy = tap / 3 - 1, dx = tap % 3 - 1;
    const int sy = py + dy, sx = px + dx;
    const bool inb = (sy >= 0) && (sy < 128) && (sx >= 0) && (sx < 128);
    const int srec = (b << 14) + sy * 128 + sx;
#pragma unroll
    for (int kc = 0; kc < 2; ++kc) {
      v16bf bm = {};
      if (inb) bm = *(const v16bf*)(compT + srec * 64 + kc * 32 + half * 16);
#pragma unroll
      for (int mt = 0; mt < 7; ++mt) {
        const __bf16* wrow = wA2 + (tap * 112 + mt * 16 + m) * 64 + kc * 32;
        const v8bf a0 = *(const v8bf*)(wrow + half * 16);
        const v8bf a1 = *(const v8bf*)(wrow + half * 16 + 8);
        v16bf a;
#pragma unroll
        for (int e = 0; e < 8; ++e) { a[e] = a0[e]; a[8 + e] = a1[e]; }
        acc[mt] = __builtin_amdgcn_wmma_f32_16x16x32_bf16(false, a, false, bm,
                                                          (short)0, acc[mt],
                                                          false, false);
      }
    }
  }
  const int rec = (b << 14) + p0 + m;
#pragma unroll
  for (int mt = 0; mt < 7; ++mt) {
    v8bf o;
#pragma unroll
    for (int r = 0; r < 8; ++r) {
      const int co = mt * 16 + half * 8 + r;
      float v = 0.0f;
      if (co < 100) {
        const float inv = g[co] * __frsqrt_rn(va[co] + BN_EPS);
        v = fmaf(acc[mt][r] + bias[co], inv, be[co] - mu[co] * inv);
      }
      o[r] = (__bf16)v;
    }
    *(v8bf*)(encT + rec * 112 + mt * 16 + half * 8) = o;  // 16B store
  }
}

// ---------------------------------------------------------------------------
// Kernel 3: pixel-shuffle(2) + softmax over 25 taps. The 25-tap gather now
// reads one 224B record (stride 8B) instead of 25 planes 16K apart.
// ---------------------------------------------------------------------------
__global__ __launch_bounds__(256) void k_shuffle_softmax(
    const __bf16* __restrict__ encT, __bf16* __restrict__ wm)
{
  const int idx = blockIdx.x * 256 + threadIdx.x;  // 4 * 65536 pixels
  const int b = idx >> 16, pix = idx & 65535;
  const int y = pix >> 8, x = pix & 255;
  const int sub = (y & 1) * 2 + (x & 1);           // pixel-shuffle sub-channel
  const int rec = (b << 14) + (y >> 1) * 128 + (x >> 1);

  float v[25], mx = -1e30f;
#pragma unroll
  for (int k = 0; k < 25; ++k) {
    v[k] = (float)encT[rec * 112 + k * 4 + sub];
    mx = fmaxf(mx, v[k]);
  }
  float s = 0.0f;
#pragma unroll
  for (int k = 0; k < 25; ++k) { v[k] = __expf(v[k] - mx); s += v[k]; }
  const float inv = __frcp_rn(s);
#pragma unroll
  for (int k = 0; k < 25; ++k)
    wm[((b * 25 + k) << 16) + pix] = (__bf16)(v[k] * inv);
}

// ---------------------------------------------------------------------------
// Kernel 4: CARAFE apply. Block = 8x32 output tile; stage the 8x20 input halo
// for all 64 channels in 40 KB LDS, then 25 FMAs/channel/pixel from LDS.
// ---------------------------------------------------------------------------
__global__ __launch_bounds__(256) void k_apply(
    const float* __restrict__ X, const __bf16* __restrict__ wm,
    float* __restrict__ out)
{
  __shared__ float xs[64 * 160];       // 64 ch x (8 rows x 20 cols) = 40 KB
  const int b    = blockIdx.x >> 8;
  const int tile = blockIdx.x & 255;
  const int y0 = (tile >> 3) << 3;     // 32 tile-rows of 8
  const int x0 = (tile & 7) << 5;      // 8 tile-cols of 32
  const int ty = threadIdx.x >> 5, tx = threadIdx.x & 31;
  const int y = y0 + ty, x = x0 + tx;

  float wreg[25];
#pragma unroll
  for (int k = 0; k < 25; ++k)
    wreg[k] = (float)wm[((b * 25 + k) << 16) + (y << 8) + x];

  const int r0 = (y0 >> 1) - 2, c0 = (x0 >> 1) - 2;
  for (int t = threadIdx.x; t < 64 * 160; t += 256) {
    const int ch = t / 160, rem = t - ch * 160;
    const int rr = rem / 20, cc = rem - rr * 20;
    const int iy = r0 + rr, ix = c0 + cc;
    xs[t] = (iy >= 0 && iy < 128 && ix >= 0 && ix < 128)
                ? X[((b * 64 + ch) << 14) + iy * 128 + ix]
                : 0.0f;
  }
  __syncthreads();

  int off[25];                         // tap (i,j): row (ty+2i)>>1, col (tx+2j)>>1
#pragma unroll
  for (int i = 0; i < 5; ++i)
#pragma unroll
    for (int j = 0; j < 5; ++j)
      off[i * 5 + j] = ((ty + 2 * i) >> 1) * 20 + ((tx + 2 * j) >> 1);

  for (int c = 0; c < 64; ++c) {
    const float* base = xs + c * 160;
    float acc = 0.0f;
#pragma unroll
    for (int k = 0; k < 25; ++k) acc = fmaf(wreg[k], base[off[k]], acc);
    out[((b * 64 + c) << 16) + (y << 8) + x] = acc;
  }
}

// ---------------------------------------------------------------------------
extern "C" void kernel_launch(void* const* d_in, const int* in_sizes, int n_in,
                              void* d_out, int out_size, void* d_ws, size_t ws_size,
                              hipStream_t stream) {
  const float* X          = (const float*)d_in[0];
  const float* comp_w     = (const float*)d_in[1];
  const float* comp_gamma = (const float*)d_in[2];
  const float* comp_beta  = (const float*)d_in[3];
  const float* comp_mean  = (const float*)d_in[4];
  const float* comp_var   = (const float*)d_in[5];
  const float* enc_w      = (const float*)d_in[6];
  const float* enc_b      = (const float*)d_in[7];
  const float* enc_gamma  = (const float*)d_in[8];
  const float* enc_beta   = (const float*)d_in[9];
  const float* enc_mean   = (const float*)d_in[10];
  const float* enc_var    = (const float*)d_in[11];
  float* out = (float*)d_out;

  // Workspace layout (bf16 intermediates, ~44.7 MB)
  char* p = (char*)d_ws;
  __bf16* Xt    = (__bf16*)p;  p += 4ull * 16384 * 64  * 2;   //  8.0 MB
  __bf16* compT = (__bf16*)p;  p += 4ull * 16384 * 64  * 2;   //  8.0 MB
  __bf16* encT  = (__bf16*)p;  p += 4ull * 16384 * 112 * 2;   // 14.0 MB
  __bf16* wmb   = (__bf16*)p;  p += 4ull * 25 * 65536  * 2;   // 12.5 MB
  __bf16* wA1   = (__bf16*)p;  p += 4096ull * 2;
  __bf16* wA2   = (__bf16*)p;

  // 0) weight repack (68608 elems) + X transpose to pixel-major bf16
  k_repack<<<268, 256, 0, stream>>>(comp_w, enc_w, wA1, wA2);
  k_transposeX<<<1024, 256, 0, stream>>>(X, Xt);
  // 1) 1x1 conv + BN + ReLU: 4096 waves (4b x 1024pt), 8 WMMAs/wave
  k_compress<<<512, 256, 0, stream>>>(Xt, wA1, comp_gamma, comp_beta,
                                      comp_mean, comp_var, compT);
  // 2) 3x3 conv + bias + BN: 4096 waves, 126 WMMAs/wave
  k_encode<<<512, 256, 0, stream>>>(compT, wA2, enc_b, enc_gamma, enc_beta,
                                    enc_mean, enc_var, encT);
  // 3) pixel shuffle + softmax over 25 taps: 262144 pixels
  k_shuffle_softmax<<<1024, 256, 0, stream>>>(encT, wmb);
  // 4) CARAFE apply: 4b x 256 tiles of 8x32
  k_apply<<<1024, 256, 0, stream>>>(X, wmb, out);
}
